// BasicSelfAttention_24816321036808
// MI455X (gfx1250) — matrix operand order repaired and checked by reading, not compiled, not run on hardware
//
#include <hip/hip_runtime.h>
#include <hip/hip_bf16.h>

// ---------------------------------------------------------------------------
// Problem constants (reference): B=8, N=256, T=48, C=320, heads 2/4/4, HD=32
// ---------------------------------------------------------------------------
#define BB 8
#define NN 256
#define TT 48
#define CC 320
#define HD 32
#define MROWS ((long)BB * NN * TT)   // 98304
#define PCOLS 832                    // q(64)|tq(128)|tk(128)|tv(128)|gq(128)|gk(128)|gv(128)
#define SCALE 0.17677669529663687f   // 32^-0.5

typedef _Float16 f16;
typedef __attribute__((ext_vector_type(16))) _Float16 v16h;
typedef __attribute__((ext_vector_type(8)))  _Float16 v8h;
typedef __attribute__((ext_vector_type(8)))  float    v8f;
typedef __attribute__((ext_vector_type(8)))  float    v8fx;

// v_wmma_f32_16x16x32_f16: D = A(16x32) * B(32x16) + C
__device__ __forceinline__ v8f wmma16(v16h a, v16h b, v8f c) {
  return __builtin_amdgcn_wmma_f32_16x16x32_f16(
      /*neg_a=*/false, a, /*neg_b=*/false, b,
      /*c_mod=*/(short)0, c, /*reuse_a=*/false, /*reuse_b=*/false);
}

// gfx1250 async global->LDS copy (16B per lane), tracked by ASYNCcnt.
// Low 32 bits of a generic LDS pointer are the wave-relative LDS byte address.
__device__ __forceinline__ void async_ld16(void* lds, const void* gaddr) {
  unsigned l = (unsigned)(unsigned long long)lds;
  asm volatile("global_load_async_to_lds_b128 %0, %1, off"
               :: "v"(l), "v"(gaddr) : "memory");
}
__device__ __forceinline__ void async_wait0() {
  asm volatile("s_wait_asynccnt 0x0" ::: "memory");
}

// 16-bit A-matrix fragment: lane (r,hi) holds row r, 16 contiguous K
// elements [hi*16, hi*16+16) (matches B fragment order).
__device__ __forceinline__ v16h ld_a_f16(const f16* __restrict__ rowp, int hi) {
  v8h lo = *(const v8h*)(rowp + hi * 16);
  v8h hh = *(const v8h*)(rowp + hi * 16 + 8);
  v16h a;
#pragma unroll
  for (int e = 0; e < 8; ++e) { a[e] = lo[e]; a[8 + e] = hh[e]; }
  return a;
}

// B-matrix fragment: lane (c,hi) needs 16 contiguous f16 starting at K=16*hi
// of column (c) -> caller passes &Bst[col][16*hi]-style pointer.
__device__ __forceinline__ v16h ld_b_f16(const f16* __restrict__ p) {
  v8h lo = *(const v8h*)p;
  v8h hh = *(const v8h*)(p + 8);
  v16h b;
#pragma unroll
  for (int e = 0; e < 8; ++e) { b[e] = lo[e]; b[8 + e] = hh[e]; }
  return b;
}

template <typename AT>
__device__ __forceinline__ v16h load_a(const AT* __restrict__ arow, int hi) {
  if constexpr (sizeof(AT) == 2) {
    return ld_a_f16((const f16*)arow, hi);
  } else {
    v8fx lo = *(const v8fx*)(arow + hi * 16);
    v8fx hh = *(const v8fx*)(arow + hi * 16 + 8);
    v16h a;
#pragma unroll
    for (int e = 0; e < 8; ++e) { a[e] = (f16)lo[e]; a[8 + e] = (f16)hh[e]; }
    return a;
  }
}

// ---------------------------------------------------------------------------
// f32 -> f16 flat convert (also packs weights: row-major [O,K] stays
// row-major; concatenation = destination offset).
// ---------------------------------------------------------------------------
__global__ void cvt_f16(const float* __restrict__ src, f16* __restrict__ dst,
                        long n) {
  long i = ((long)blockIdx.x * 256 + threadIdx.x) * 4;
  if (i >= n) return;
  const float4 v = *(const float4*)(src + i);
  dst[i + 0] = (f16)v.x;
  dst[i + 1] = (f16)v.y;
  dst[i + 2] = (f16)v.z;
  dst[i + 3] = (f16)v.w;
}

// ---------------------------------------------------------------------------
// WMMA GEMM: C[M,N] = A[M,K] * W^T + bias, W given f16 row-major [N,K].
// Block = 256 threads (8 waves) -> 128x64 output tile; wave w owns rows
// [16w,16w+16) x 64 cols (4 accumulators; A regs reused 4x). B panel (64x32
// f16) staged to LDS per K-chunk via async b128 copies, double-buffered.
// M % 128 == 0, N % 64 == 0, K % 32 == 0.
// ---------------------------------------------------------------------------
template <typename AT, typename OT>
__global__ __launch_bounds__(256) void gemm_wmma(
    const AT* __restrict__ A, const f16* __restrict__ Bw,
    const float* __restrict__ bias, OT* __restrict__ Cm,
    long M, int N, int K, int bias_len) {
  __shared__ __attribute__((aligned(16))) f16 Bst[2][64][32];

  int  nb = N >> 6;
  int  nblk = blockIdx.x % nb;
  long mblk = blockIdx.x / nb;
  int  tid = threadIdx.x;
  int  w = tid >> 5, lane = tid & 31;
  int  c = lane & 15, hi = lane >> 4;
  long row = mblk * 128 + w * 16 + c;   // A row this lane streams
  int  c0 = nblk << 6;

  // stage K-chunk 0: thread t -> col c0 + t/4, 8 f16 at (t%4)*8
  {
    int col = tid >> 2, kk = (tid & 3) * 8;
    async_ld16(&Bst[0][col][kk], Bw + (long)(c0 + col) * K + kk);
  }

  v8f acc[4] = {{}, {}, {}, {}};
  const int nk = K >> 5;
  for (int kc = 0; kc < nk; ++kc) {
    async_wait0();
    __syncthreads();
    int buf = kc & 1;
    if (kc + 1 < nk) {  // prefetch next chunk into other buffer
      int col = tid >> 2, kk = (tid & 3) * 8;
      async_ld16(&Bst[buf ^ 1][col][kk],
                 Bw + (long)(c0 + col) * K + (kc + 1) * 32 + kk);
    }
    v16h a = load_a<AT>(A + row * K + kc * 32, hi);
#pragma unroll
    for (int ct = 0; ct < 4; ++ct) {
      v16h b = ld_b_f16(&Bst[buf][ct * 16 + c][16 * hi]);
      acc[ct] = wmma16(a, b, acc[ct]);
    }
  }

#pragma unroll
  for (int ct = 0; ct < 4; ++ct) {
    int col = c0 + ct * 16 + c;
    float bv = (col < bias_len) ? bias[col] : 0.0f;
#pragma unroll
    for (int i = 0; i < 8; ++i) {
      long r = mblk * 128 + w * 16 + hi * 8 + i;
      Cm[r * N + col] = (OT)(acc[ct][i] + bv);
    }
  }
}

// ---------------------------------------------------------------------------
// branch0 (TEMPORAL=false, 2 heads) + temporal (TEMPORAL=true, 4 heads).
// One block per (b,n,head); 3 waves, wave w owns query rows [16w,16w+16).
// Temporal adds pos[b,n,ht,k,q] to scaled scores. Output -> concat cols 0..191.
// ---------------------------------------------------------------------------
template <bool TEMPORAL>
__global__ __launch_bounds__(96) void local_attn(
    const f16* __restrict__ proj, const f16* __restrict__ kbuf,
    const f16* __restrict__ vbuf, const float* __restrict__ pos,
    f16* __restrict__ cat) {
  int blk = blockIdx.x;
  const int nh = TEMPORAL ? 4 : 2;
  int h  = blk % nh;
  int bn = blk / nh;
  int n  = bn % NN;
  int b  = bn / NN;

  __shared__ __attribute__((aligned(16))) f16  Qh[48][32];
  __shared__ __attribute__((aligned(16))) f16  Kh[64][32];  // rows 48..63 zero
  __shared__ __attribute__((aligned(16))) f16  Vh[64][32];
  __shared__ float Sb[3][16][48];
  __shared__ __attribute__((aligned(16))) f16  Pb[3][16][64];

  int tid = threadIdx.x;
  long rowbase = ((long)(b * NN + n)) * TT;

  const int qc = TEMPORAL ? (64 + h * 32) : (h * 32);
  const int kc = TEMPORAL ? (192 + h * 32) : (h * 32);
  const int vc = TEMPORAL ? (320 + h * 32) : (h * 32);
  for (int i = tid; i < 48 * 32; i += 96) {
    int t = i >> 5, d = i & 31;
    long r = rowbase + t;
    Qh[t][d] = proj[r * PCOLS + qc + d];
    if (TEMPORAL) {
      Kh[t][d] = proj[r * PCOLS + kc + d];
      Vh[t][d] = proj[r * PCOLS + vc + d];
    } else {
      Kh[t][d] = kbuf[r * 64 + kc + d];
      Vh[t][d] = vbuf[r * 64 + vc + d];
    }
  }
  for (int i = tid; i < 16 * 32; i += 96) {
    int t = 48 + (i >> 5), d = i & 31;
    Kh[t][d] = (f16)0.0f;
    Vh[t][d] = (f16)0.0f;
  }
  __syncthreads();

  int w = tid >> 5, lane = tid & 31;
  int c = lane & 15, hi = lane >> 4;
  int r0 = w * 16;

  // ---- scores S[16,48] = Q strip x K^T (3 WMMAs, K-dim = HD = 32) ----
  v16h aQ = ld_a_f16(&Qh[r0 + c][0], hi);
#pragma unroll
  for (int kt = 0; kt < 3; ++kt) {
    v16h bK = ld_b_f16(&Kh[kt * 16 + c][16 * hi]);
    v8f z = {};
    v8f s = wmma16(aQ, bK, z);
    float pv[8];
    if (TEMPORAL) {
      // st += swapaxes(pos,-2,-1): add pos[b,n,h,k,q]; q contiguous -> batched
      const float* pbase =
          pos + ((((long)(b * NN + n)) * 4 + h) * TT + (kt * 16 + c)) * TT +
          r0 + hi * 8;
#pragma unroll
      for (int i = 0; i < 8; ++i) pv[i] = pbase[i];
    }
#pragma unroll
    for (int i = 0; i < 8; ++i) {
      float v = s[i] * SCALE;
      if (TEMPORAL) v += pv[i];
      Sb[w][hi * 8 + i][kt * 16 + c] = v;
    }
  }
  __syncthreads();

  // ---- row softmax over 48 keys; probs f16, K padded to 64 ----
  if (lane < 16) {
    int r = lane;
    float mx = -3.0e38f;
#pragma unroll
    for (int k = 0; k < 48; ++k) mx = fmaxf(mx, Sb[w][r][k]);
    float sum = 0.0f;
#pragma unroll
    for (int k = 0; k < 48; ++k) {
      float p = __expf(Sb[w][r][k] - mx);
      Sb[w][r][k] = p;
      sum += p;
    }
    float inv = 1.0f / sum;
#pragma unroll
    for (int k = 0; k < 48; ++k) Pb[w][r][k] = (f16)(Sb[w][r][k] * inv);
#pragma unroll
    for (int k = 48; k < 64; ++k) Pb[w][r][k] = (f16)0.0f;
  }
  __syncthreads();

  // ---- out[16,32] = P[16,64] x V[64,32]: 2 K-chunks x 2 d-tiles ----
  v8f o0 = {}, o1 = {};
#pragma unroll
  for (int kk = 0; kk < 2; ++kk) {
    v16h aP = ld_a_f16(&Pb[w][c][kk * 32], hi);
    v16h b0, b1;
#pragma unroll
    for (int e = 0; e < 16; ++e) {
      int key = kk * 32 + 16 * hi + e;
      b0[e] = Vh[key][c];
      b1[e] = Vh[key][16 + c];
    }
    o0 = wmma16(aP, b0, o0);
    o1 = wmma16(aP, b1, o1);
  }

  const int ocol = TEMPORAL ? (64 + h * 32) : (h * 32);
#pragma unroll
  for (int i = 0; i < 8; ++i) {
    int tq = r0 + hi * 8 + i;
    cat[(rowbase + tq) * CC + ocol + c]      = (f16)o0[i];
    cat[(rowbase + tq) * CC + ocol + 16 + c] = (f16)o1[i];
  }
}

// ---------------------------------------------------------------------------
// Geo attention over N=256 with boolean mask. One block per (b,t,head);
// 8 waves, each wave does 2 query tiles of 16 rows, flash-style online softmax
// over 4 key chunks of 64. K/V staged to LDS via async b128 copies.
// Cross-lane reductions batched step-outer/row-inner so all 8 rows' bpermutes
// issue under one DS wait per butterfly step. Output written with the
// reference's raw reshape: (b,t,n) -> concat flat row b*12288 + t*256 + n.
// ---------------------------------------------------------------------------
__global__ __launch_bounds__(256) void geo_attn(
    const f16* __restrict__ proj, const unsigned char* __restrict__ mask,
    f16* __restrict__ cat) {
  int blk = blockIdx.x;
  int h  = blk & 3;
  int bt = blk >> 2;
  int t  = bt % TT;
  int b  = bt / TT;

  __shared__ __attribute__((aligned(16))) f16 Kh[256][32];
  __shared__ __attribute__((aligned(16))) f16 Vh[256][32];
  __shared__ __attribute__((aligned(16))) f16 Pb[8][16][64];

  int tid = threadIdx.x;
  int wave = tid >> 5, lane = tid & 31;
  int c = lane & 15, hi = lane >> 4;

  // stage K (gk, cols 576+) and V (gv, cols 704+): one row per thread, async
  {
    int nn = tid;
    long r = ((long)(b * NN + nn)) * TT + t;
    const f16* ksrc = proj + r * PCOLS + 576 + h * 32;
    const f16* vsrc = proj + r * PCOLS + 704 + h * 32;
#pragma unroll
    for (int j = 0; j < 4; ++j) {
      async_ld16(&Kh[nn][j * 8], ksrc + j * 8);
      async_ld16(&Vh[nn][j * 8], vsrc + j * 8);
    }
  }
  async_wait0();
  __syncthreads();

  for (int rep = 0; rep < 2; ++rep) {
    int qtile = wave + rep * 8;

    // Q tile in A layout straight from global (gq, cols 448+)
    int  qrow = qtile * 16 + c;
    long qr = ((long)(b * NN + qrow)) * TT + t;
    v16h aQ = ld_a_f16(proj + qr * PCOLS + 448 + h * 32, hi);

    float m[8], l[8];
#pragma unroll
    for (int i = 0; i < 8; ++i) { m[i] = -3.0e38f; l[i] = 0.0f; }
    v8f o0 = {}, o1 = {};

    for (int kc = 0; kc < 4; ++kc) {
      // 4 score subtiles [16x16], K-dim = HD = 32
      v8f s[4];
#pragma unroll
      for (int st = 0; st < 4; ++st) {
        v16h bK = ld_b_f16(&Kh[kc * 64 + st * 16 + c][16 * hi]);
        v8f z = {};
        s[st] = wmma16(aQ, bK, z);
      }

      // mask bytes for all 8 rows x 4 subtiles: one big load clause
      unsigned char mb[8][4];
#pragma unroll
      for (int i = 0; i < 8; ++i) {
        const unsigned char* mrow =
            mask + (qtile * 16 + hi * 8 + i) * NN + kc * 64;
#pragma unroll
        for (int st = 0; st < 4; ++st) mb[i][st] = mrow[st * 16 + c];
      }

      // masked scaled scores + per-lane partial row max
      float pm[8];
#pragma unroll
      for (int i = 0; i < 8; ++i) {
        float mx = -3.0e38f;
#pragma unroll
        for (int st = 0; st < 4; ++st) {
          float v = s[st][i] * SCALE;
          if (mb[i][st]) v = -3.0e38f;
          s[st][i] = v;
          mx = fmaxf(mx, v);
        }
        pm[i] = mx;
      }
      // batched butterfly max across the 16 col lanes (rows independent)
#pragma unroll
      for (int off = 1; off < 16; off <<= 1) {
        float tmp[8];
#pragma unroll
        for (int i = 0; i < 8; ++i) tmp[i] = __shfl_xor(pm[i], off, 32);
#pragma unroll
        for (int i = 0; i < 8; ++i) pm[i] = fmaxf(pm[i], tmp[i]);
      }

      // online rescale + exp + per-lane partial row sum
      float rs[8];
#pragma unroll
      for (int i = 0; i < 8; ++i) {
        float mn = fmaxf(m[i], pm[i]);
        float corr = __expf(m[i] - mn);
        o0[i] *= corr;
        o1[i] *= corr;
        float r = 0.0f;
#pragma unroll
        for (int st = 0; st < 4; ++st) {
          float p = __expf(s[st][i] - mn);
          s[st][i] = p;
          r += p;
        }
        rs[i] = r;
        l[i] *= corr;
        m[i] = mn;
      }
      // batched butterfly sum
#pragma unroll
      for (int off = 1; off < 16; off <<= 1) {
        float tmp[8];
#pragma unroll
        for (int i = 0; i < 8; ++i) tmp[i] = __shfl_xor(rs[i], off, 32);
#pragma unroll
        for (int i = 0; i < 8; ++i) rs[i] += tmp[i];
      }
#pragma unroll
      for (int i = 0; i < 8; ++i) l[i] += rs[i];

      // stage probs (C layout -> LDS -> A layout)
#pragma unroll
      for (int i = 0; i < 8; ++i)
#pragma unroll
        for (int st = 0; st < 4; ++st)
          Pb[wave][hi * 8 + i][st * 16 + c] = (f16)s[st][i];
      __syncthreads();

      // AV: P[16,64] x V[64,32] -> 2 K-chunks x 2 d-tiles
#pragma unroll
      for (int kk = 0; kk < 2; ++kk) {
        v16h aP = ld_a_f16(&Pb[wave][c][kk * 32], hi);
        v16h b0, b1;
#pragma unroll
        for (int e = 0; e < 16; ++e) {
          int key = kc * 64 + kk * 32 + 16 * hi + e;
          b0[e] = Vh[key][c];
          b1[e] = Vh[key][16 + c];
        }
        o0 = wmma16(aP, b0, o0);
        o1 = wmma16(aP, b1, o1);
      }
      __syncthreads();
    }

    // normalize + store with raw-reshape remap
#pragma unroll
    for (int i = 0; i < 8; ++i) {
      float inv = 1.0f / l[i];
      int qn = qtile * 16 + hi * 8 + i;
      long orow = (long)b * (TT * NN) + (long)t * NN + qn;
      cat[orow * CC + 192 + h * 32 + c]      = (f16)(o0[i] * inv);
      cat[orow * CC + 192 + h * 32 + 16 + c] = (f16)(o1[i] * inv);
    }
  }
}

// ---------------------------------------------------------------------------
// Host launch
// ---------------------------------------------------------------------------
extern "C" void kernel_launch(void* const* d_in, const int* in_sizes, int n_in,
                              void* d_out, int out_size, void* d_ws, size_t ws_size,
                              hipStream_t stream) {
  (void)in_sizes; (void)n_in; (void)out_size; (void)ws_size;
  const float* query = (const float*)d_in[0];
  const float* key   = (const float*)d_in[1];
  const float* value = (const float*)d_in[2];
  const float* pos   = (const float*)d_in[3];
  const float* Wq  = (const float*)d_in[4];  const float* bq = (const float*)d_in[5];
  const float* Wk  = (const float*)d_in[6];  const float* bk = (const float*)d_in[7];
  const float* Wv  = (const float*)d_in[8];  const float* bv = (const float*)d_in[9];
  const float* Wtq = (const float*)d_in[10];
  const float* Wtk = (const float*)d_in[11];
  const float* Wtv = (const float*)d_in[12];
  const float* Wgq = (const float*)d_in[13];
  const float* Wgk = (const float*)d_in[14];
  const float* Wgv = (const float*)d_in[15];
  const float* Wo  = (const float*)d_in[16]; const float* bo = (const float*)d_in[17];
  const unsigned char* mask = (const unsigned char*)d_in[18];  // bool [N,N]
  float* out = (float*)d_out;

  char* ws = (char*)d_ws;
  size_t off = 0;
  auto take = [&](size_t bytes) -> char* {
    char* p = ws + off;
    off += (bytes + 255) & ~(size_t)255;
    return p;
  };
  f16* proj = (f16*)take(MROWS * PCOLS * sizeof(f16));   // q|tq|tk|tv|gq|gk|gv
  f16* kbuf = (f16*)take(MROWS * 64 * sizeof(f16));
  f16* vbuf = (f16*)take(MROWS * 64 * sizeof(f16));
  f16* cat  = (f16*)take(MROWS * CC * sizeof(f16));      // concat buffer
  f16* qf16 = (f16*)take(MROWS * CC * sizeof(f16));      // query in f16
  f16* Wcat = (f16*)take((size_t)PCOLS * CC * sizeof(f16));  // [832][320]
  f16* Wkf  = (f16*)take((size_t)64 * CC * sizeof(f16));
  f16* Wvf  = (f16*)take((size_t)64 * CC * sizeof(f16));
  f16* Wof  = (f16*)take((size_t)CC * CC * sizeof(f16));

  auto cvt = [&](const float* src, f16* dst, long n) {
    cvt_f16<<<(int)((n / 4 + 255) / 256), 256, 0, stream>>>(src, dst, n);
  };
  // pack weights: plain f16 convert, row-major [O,K]; concat = row offsets
  cvt(Wq,  Wcat + (long)0   * CC, 64  * CC);
  cvt(Wtq, Wcat + (long)64  * CC, 128 * CC);
  cvt(Wtk, Wcat + (long)192 * CC, 128 * CC);
  cvt(Wtv, Wcat + (long)320 * CC, 128 * CC);
  cvt(Wgq, Wcat + (long)448 * CC, 128 * CC);
  cvt(Wgk, Wcat + (long)576 * CC, 128 * CC);
  cvt(Wgv, Wcat + (long)704 * CC, 128 * CC);
  cvt(Wk, Wkf, 64 * CC);
  cvt(Wv, Wvf, 64 * CC);
  cvt(Wo, Wof, (long)CC * CC);
  cvt(query, qf16, MROWS * CC);   // A-side f16 (re-read 13x by proj GEMM)

  const int mb = (int)(MROWS / 128);  // 768
  // fused query-side projection: [98304,320] x [320,832] (+bq on cols 0..63)
  gemm_wmma<f16, f16><<<mb * (PCOLS / 64), 256, 0, stream>>>(
      qf16, Wcat, bq, proj, MROWS, PCOLS, CC, 64);
  // k / v projections for branch 0 (single 64-wide strip: read f32 A once)
  gemm_wmma<float, f16><<<mb, 256, 0, stream>>>(
      key, Wkf, bk, kbuf, MROWS, 64, CC, 64);
  gemm_wmma<float, f16><<<mb, 256, 0, stream>>>(
      value, Wvf, bv, vbuf, MROWS, 64, CC, 64);

  // attention branches -> concat buffer
  local_attn<false><<<BB * NN * 2, 96, 0, stream>>>(proj, kbuf, vbuf, pos, cat);
  local_attn<true><<<BB * NN * 4, 96, 0, stream>>>(proj, kbuf, vbuf, pos, cat);
  geo_attn<<<BB * TT * 4, 256, 0, stream>>>(proj, mask, cat);

  // output projection: [98304,320] x [320,320] + bo -> f32 out
  gemm_wmma<f16, float><<<mb * (CC / 64), 256, 0, stream>>>(
      cat, Wof, bo, out, MROWS, CC, CC, CC);
}
